// PaperContextGNN_72353019069072
// MI455X (gfx1250) — compile-verified, matrix operand order checked by reading, not run on hardware
//
#include <hip/hip_runtime.h>

#define D_IN   1024
#define D_HID  512
#define D_B    256
#define NLAY   3

typedef __attribute__((ext_vector_type(16))) __bf16 v16bf;
typedef __attribute__((ext_vector_type(8)))  float  v8f;
typedef unsigned int v4u __attribute__((ext_vector_type(4)));
typedef int          v4i __attribute__((ext_vector_type(4)));
typedef int          v8i __attribute__((ext_vector_type(8)));

__device__ __forceinline__ unsigned short f2bf(float f) {
  unsigned int u = __float_as_uint(f);
  u += 0x7FFFu + ((u >> 16) & 1u);          // round-to-nearest-even
  return (unsigned short)(u >> 16);
}

// ---------------------------------------------------------------------------
// TDM: 2D tile load (bf16 elements) global -> LDS, one instruction per tile.
// D# layout per CDNA5 ISA 8.3/8.4:
//   group0: [1:0]=count=1, [63:32]=lds_addr, [120:64]=global_addr, [127:126]=type=2
//   group1: data_size=1(2B) @ [17:16]; tensor_dim0 @ [79:48]; tensor_dim1 @ [111:80];
//           tile_dim0 @ [127:112]; tile_dim1 @ [143:128]; tensor_dim0_stride @ [207:160]
// Tensor dims are the remaining extent from the tile origin so OOB rows zero-fill.
// This toolchain exposes the 6-arg builtin: (v4u, v8i, v4i, v4i, v8i, i32 cpol).
// ---------------------------------------------------------------------------
__device__ __forceinline__ void tdm_load_2d_bf16(
    const unsigned short* gptr, unsigned lds_off,
    int tensor_d0, int tensor_d1, int tile_d0, int tile_d1,
    long long row_stride_elems)
{
  unsigned long long ga = (unsigned long long)(uintptr_t)gptr;
  v4u g0;
  g0[0] = 1u;                                            // count=1, user mode
  g0[1] = lds_off;                                       // lds_addr (bytes)
  g0[2] = (unsigned)(ga & 0xFFFFFFFFull);                // global_addr[31:0]
  g0[3] = (unsigned)((ga >> 32) & 0x1FFFFFFull) | (2u << 30);  // addr[56:32] | type=2
  v8i g1;
  g1[0] = (int)(1u << 16);                               // data_size = 2 bytes
  g1[1] = (int)((unsigned)tensor_d0 << 16);              // tensor_dim0[15:0]
  g1[2] = (int)(((unsigned)tensor_d0 >> 16) | ((unsigned)tensor_d1 << 16));
  g1[3] = (int)(((unsigned)tensor_d1 >> 16) | ((unsigned)tile_d0 << 16));
  g1[4] = tile_d1 & 0xFFFF;                              // tile_dim1, tile_dim2=0
  g1[5] = (int)(unsigned)(row_stride_elems & 0xFFFFFFFFll);
  g1[6] = (int)(unsigned)((row_stride_elems >> 32) & 0xFFFFll);
  g1[7] = 0;
  v4i gz4 = {0, 0, 0, 0};                                // groups 2/3 unused (2D)
  v8i gz8 = {0, 0, 0, 0, 0, 0, 0, 0};
  __builtin_amdgcn_tensor_load_to_lds(g0, g1, gz4, gz4, gz8, 0);
}

// ---------------------------------------------------------------------------
// Tiled bf16 WMMA GEMM:  C[M,Nc] = act(A[M,K] @ B[K,Nc] + bias)
// Block: 256 threads = 8 waves.  Tile: 128(M) x 64(N) x 64(K).
// A tile staged by the Tensor Data Mover; B tile staged transposed by all
// threads.  Each wave: 16x64 strip, 8 x v_wmma_f32_16x16x32_bf16 per K-step.
// ---------------------------------------------------------------------------
#define BM 128
#define BN 64
#define BK 64

union Frag16 { v16bf v; unsigned int d[8]; };

__global__ __launch_bounds__(256) void gemm_bf16_wmma(
    const unsigned short* __restrict__ A,   // [M,K]  row-major bf16 bits
    const unsigned short* __restrict__ B,   // [K,Nc] row-major bf16 bits
    const float* __restrict__ bias,         // [Nc]
    float* __restrict__ Cf,                 // fp32 out (or nullptr)
    unsigned short* __restrict__ Cb,        // bf16 out (or nullptr)
    int M, int K, int Nc, int relu)
{
  __shared__ unsigned short sA[BM * BK];          // 16 KB, row-major [m][k]
  __shared__ unsigned short sB[BN * BK];          // 8 KB, transposed [n][k]

  const int tid  = threadIdx.x;
  const int wv   = tid >> 5;
  const int lane = tid & 31;
  const int hlf  = lane >> 4;                     // lane group (0: 0-15, 1: 16-31)
  const int l    = lane & 15;
  const int bm   = blockIdx.x * BM;
  const int bn   = blockIdx.y * BN;
  const unsigned sA_off = (unsigned)(uintptr_t)&sA[0];   // LDS byte offset

  v8f acc[4] = {};

  for (int k0 = 0; k0 < K; k0 += BK) {
    // ---- A tile (128x64 bf16) via Tensor Data Mover (wave 0 issues once)
    if (wv == 0) {
      tdm_load_2d_bf16(A + (size_t)bm * K + k0, sA_off,
                       K - k0, M - bm, BK, BM, (long long)K);
    }
    // ---- B tile (64x64 bf16) transposed into sB[n*BK+k]; all 256 threads
    #pragma unroll
    for (int it = 0; it < 2; ++it) {
      int i  = tid + it * 256;           // 512 uint4 total
      int kr = i >> 3;                   // 8 uint4 per 64-elem row
      int n8 = (i & 7) * 8;
      uint4 v = *(const uint4*)(B + (size_t)(k0 + kr) * Nc + bn + n8);
      const unsigned short* pv = (const unsigned short*)&v;
      #pragma unroll
      for (int j = 0; j < 8; ++j)
        sB[(n8 + j) * BK + kr] = pv[j];
    }
    if (wv == 0) __builtin_amdgcn_s_wait_tensorcnt(0);
    __syncthreads();

    // ---- two K-halves of 32; fragment layout per ISA 7.12.2 (16-bit A 16x32):
    // VGPR r holds K pair base (r<4 ? 2r : 16+2(r-4)) + 8*laneGroup
    #pragma unroll
    for (int kh = 0; kh < 2; ++kh) {
      Frag16 af;
      const int arow = wv * 16 + l;
      #pragma unroll
      for (int r = 0; r < 8; ++r) {
        int kk = ((r < 4) ? 2 * r : 16 + 2 * (r - 4)) + hlf * 8 + kh * 32;
        af.d[r] = *(const unsigned int*)&sA[arow * BK + kk];
      }
      #pragma unroll
      for (int j = 0; j < 4; ++j) {
        Frag16 bf;
        const int brow = j * 16 + l;
        #pragma unroll
        for (int r = 0; r < 8; ++r) {
          int kk = ((r < 4) ? 2 * r : 16 + 2 * (r - 4)) + hlf * 8 + kh * 32;
          bf.d[r] = *(const unsigned int*)&sB[brow * BK + kk];
        }
        acc[j] = __builtin_amdgcn_wmma_f32_16x16x32_bf16(
            false, af.v, false, bf.v, (short)0, acc[j], false, false);
      }
    }
    __syncthreads();
  }

  // ---- epilogue: C layout VGPR i -> M = i + 8*laneGroup, N = lane&15
  #pragma unroll
  for (int j = 0; j < 4; ++j) {
    const int col = bn + j * 16 + l;
    const float bs = bias[col];
    #pragma unroll
    for (int i = 0; i < 8; ++i) {
      int row = bm + wv * 16 + hlf * 8 + i;
      if (row < M) {
        float v = acc[j][i] + bs;
        if (relu) v = fmaxf(v, 0.0f);
        if (Cf) Cf[(size_t)row * Nc + col] = v;
        if (Cb) Cb[(size_t)row * Nc + col] = f2bf(v);
      }
    }
  }
}

// ---------------------------------------------------------------------------
// LayerNorm over 1024 (in_proj front): fp32 in -> bf16 out
// ---------------------------------------------------------------------------
__global__ __launch_bounds__(256) void ln_in_kernel(
    const float* __restrict__ x, const float* __restrict__ g,
    const float* __restrict__ b, unsigned short* __restrict__ out)
{
  const int row = blockIdx.x;
  const float* xr = x + (size_t)row * D_IN;
  float vals[4], s = 0.f, sq = 0.f;
  #pragma unroll
  for (int i = 0; i < 4; ++i) {
    float v = xr[threadIdx.x + i * 256];
    vals[i] = v; s += v; sq += v * v;
  }
  #pragma unroll
  for (int o = 16; o; o >>= 1) { s += __shfl_xor(s, o, 32); sq += __shfl_xor(sq, o, 32); }
  __shared__ float ss[8], sv[8];
  int wv = threadIdx.x >> 5, ln = threadIdx.x & 31;
  if (ln == 0) { ss[wv] = s; sv[wv] = sq; }
  __syncthreads();
  if (threadIdx.x < 32) {
    s  = (ln < 8) ? ss[ln] : 0.f;
    sq = (ln < 8) ? sv[ln] : 0.f;
    #pragma unroll
    for (int o = 4; o; o >>= 1) { s += __shfl_xor(s, o, 32); sq += __shfl_xor(sq, o, 32); }
    if (ln == 0) { ss[0] = s; sv[0] = sq; }
  }
  __syncthreads();
  float mean = ss[0] * (1.f / D_IN);
  float var  = sv[0] * (1.f / D_IN) - mean * mean;
  float rstd = rsqrtf(var + 1e-5f);
  unsigned short* orow = out + (size_t)row * D_IN;
  #pragma unroll
  for (int i = 0; i < 4; ++i) {
    int c = threadIdx.x + i * 256;
    orow[c] = f2bf((vals[i] - mean) * rstd * g[c] + b[c]);
  }
}

// ---------------------------------------------------------------------------
// Final LayerNorm over 1024 of (x + delta) -> fp32 out
// ---------------------------------------------------------------------------
__global__ __launch_bounds__(256) void final_ln_kernel(
    const float* __restrict__ x, const float* __restrict__ delta,
    const float* __restrict__ g, const float* __restrict__ b,
    float* __restrict__ out)
{
  const int row = blockIdx.x;
  const float* xr = x + (size_t)row * D_IN;
  const float* dr = delta + (size_t)row * D_IN;
  float vals[4], s = 0.f, sq = 0.f;
  #pragma unroll
  for (int i = 0; i < 4; ++i) {
    int c = threadIdx.x + i * 256;
    float v = xr[c] + dr[c];
    vals[i] = v; s += v; sq += v * v;
  }
  #pragma unroll
  for (int o = 16; o; o >>= 1) { s += __shfl_xor(s, o, 32); sq += __shfl_xor(sq, o, 32); }
  __shared__ float ss[8], sv[8];
  int wv = threadIdx.x >> 5, ln = threadIdx.x & 31;
  if (ln == 0) { ss[wv] = s; sv[wv] = sq; }
  __syncthreads();
  if (threadIdx.x < 32) {
    s  = (ln < 8) ? ss[ln] : 0.f;
    sq = (ln < 8) ? sv[ln] : 0.f;
    #pragma unroll
    for (int o = 4; o; o >>= 1) { s += __shfl_xor(s, o, 32); sq += __shfl_xor(sq, o, 32); }
    if (ln == 0) { ss[0] = s; sv[0] = sq; }
  }
  __syncthreads();
  float mean = ss[0] * (1.f / D_IN);
  float var  = sv[0] * (1.f / D_IN) - mean * mean;
  float rstd = rsqrtf(var + 1e-5f);
  float* orow = out + (size_t)row * D_IN;
  #pragma unroll
  for (int i = 0; i < 4; ++i) {
    int c = threadIdx.x + i * 256;
    orow[c] = (vals[i] - mean) * rstd * g[c] + b[c];
  }
}

// ---------------------------------------------------------------------------
// LayerNorm(256) + ReLU + residual:  h += relu(LN(Z)*g + b)
// ---------------------------------------------------------------------------
__global__ __launch_bounds__(256) void ln_relu_residual_kernel(
    const float* __restrict__ Z, const float* __restrict__ g,
    const float* __restrict__ b, float* __restrict__ h)
{
  const int row = blockIdx.x;
  float v = Z[(size_t)row * D_B + threadIdx.x];
  float s = v, sq = v * v;
  #pragma unroll
  for (int o = 16; o; o >>= 1) { s += __shfl_xor(s, o, 32); sq += __shfl_xor(sq, o, 32); }
  __shared__ float ss[8], sv[8];
  int wv = threadIdx.x >> 5, ln = threadIdx.x & 31;
  if (ln == 0) { ss[wv] = s; sv[wv] = sq; }
  __syncthreads();
  if (threadIdx.x < 32) {
    s  = (ln < 8) ? ss[ln] : 0.f;
    sq = (ln < 8) ? sv[ln] : 0.f;
    #pragma unroll
    for (int o = 4; o; o >>= 1) { s += __shfl_xor(s, o, 32); sq += __shfl_xor(sq, o, 32); }
    if (ln == 0) { ss[0] = s; sv[0] = sq; }
  }
  __syncthreads();
  float mean = ss[0] * (1.f / D_B);
  float var  = sv[0] * (1.f / D_B) - mean * mean;
  float rstd = rsqrtf(var + 1e-5f);
  float z = fmaxf((v - mean) * rstd * g[threadIdx.x] + b[threadIdx.x], 0.0f);
  h[(size_t)row * D_B + threadIdx.x] += z;
}

// ---------------------------------------------------------------------------
// Scatter-add:  agg[dst[e]] += h[src[e]]   (4 edges / block, float4 reads)
// ---------------------------------------------------------------------------
__global__ __launch_bounds__(256) void scatter_add_kernel(
    const int* __restrict__ src, const int* __restrict__ dst,
    const float* __restrict__ h, float* __restrict__ agg, int E)
{
  int e = blockIdx.x * 4 + (threadIdx.x >> 6);
  if (e >= E) return;
  int t = (threadIdx.x & 63) * 4;
  int sN = src[e], dN = dst[e];
  const float4 v = *(const float4*)(h + (size_t)sN * D_B + t);
  float* ap = agg + (size_t)dN * D_B + t;
  atomicAdd(ap + 0, v.x);
  atomicAdd(ap + 1, v.y);
  atomicAdd(ap + 2, v.z);
  atomicAdd(ap + 3, v.w);
}

// ---------------------------------------------------------------------------
// Elementwise helpers
// ---------------------------------------------------------------------------
__global__ void cvt_f32_bf16_kernel(const float* __restrict__ in,
                                    unsigned short* __restrict__ out, size_t n)
{
  size_t i = (size_t)blockIdx.x * blockDim.x + threadIdx.x;
  size_t st = (size_t)gridDim.x * blockDim.x;
  for (; i < n; i += st) out[i] = f2bf(in[i]);
}

__global__ void add_cvt_bf16_kernel(const float* __restrict__ a,
                                    const float* __restrict__ b,
                                    unsigned short* __restrict__ out, size_t n)
{
  size_t i = (size_t)blockIdx.x * blockDim.x + threadIdx.x;
  size_t st = (size_t)gridDim.x * blockDim.x;
  for (; i < n; i += st) out[i] = f2bf(a[i] + b[i]);
}

// ---------------------------------------------------------------------------
extern "C" void kernel_launch(void* const* d_in, const int* in_sizes, int n_in,
                              void* d_out, int out_size, void* d_ws, size_t ws_size,
                              hipStream_t stream)
{
  const float* x     = (const float*)d_in[0];
  const int*   edge  = (const int*)d_in[1];
  const float* lng   = (const float*)d_in[2];
  const float* lnb   = (const float*)d_in[3];
  const float* W1    = (const float*)d_in[4];
  const float* b1    = (const float*)d_in[5];
  const float* W2    = (const float*)d_in[6];
  const float* b2    = (const float*)d_in[7];
  const float* Wa    = (const float*)d_in[8];
  const float* ba    = (const float*)d_in[9];
  const float* Wb    = (const float*)d_in[10];
  const float* bb    = (const float*)d_in[11];
  const float* ng    = (const float*)d_in[12];
  const float* nb    = (const float*)d_in[13];
  const float* Wd1   = (const float*)d_in[14];
  const float* bd1   = (const float*)d_in[15];
  const float* Wd2   = (const float*)d_in[16];
  const float* bd2   = (const float*)d_in[17];
  const float* outg  = (const float*)d_in[18];
  const float* outb  = (const float*)d_in[19];
  float* out = (float*)d_out;

  const int N = in_sizes[0] / D_IN;
  const int E = in_sizes[1] / 2;
  const int* src = edge;
  const int* dst = edge + E;

  // ---- workspace carve (256B aligned regions)
  char* p = (char*)d_ws;
  auto carve = [&](size_t bytes) -> char* {
    char* r = p; p += (bytes + 255) & ~(size_t)255; return r;
  };
  unsigned short* Xn  = (unsigned short*)carve((size_t)N * D_IN  * 2);
  unsigned short* H1  = (unsigned short*)carve((size_t)N * D_HID * 2);
  float*          h   = (float*)         carve((size_t)N * D_B   * 4);
  float*          agg = (float*)         carve((size_t)N * D_B   * 4);
  unsigned short* Zb  = (unsigned short*)carve((size_t)N * D_B   * 2);
  unsigned short* T1  = (unsigned short*)carve((size_t)N * D_B   * 2);
  float*          dlt = (float*)         carve((size_t)N * D_IN  * 4);
  unsigned short* W1b  = (unsigned short*)carve((size_t)D_IN  * D_HID * 2);
  unsigned short* W2b  = (unsigned short*)carve((size_t)D_HID * D_B   * 2);
  unsigned short* Wab  = (unsigned short*)carve((size_t)NLAY * D_B * D_B * 2);
  unsigned short* Wbb  = (unsigned short*)carve((size_t)NLAY * D_B * D_B * 2);
  unsigned short* Wd1b = (unsigned short*)carve((size_t)D_B   * D_HID * 2);
  unsigned short* Wd2b = (unsigned short*)carve((size_t)D_HID * D_IN  * 2);
  // safe aliases (lifetimes disjoint)
  float*          Z2 = agg;   // GIN-MLP output, written after agg consumed
  unsigned short* T2 = H1;    // delta mid [N,512], H1 free after GEMM2
  unsigned short* Hb = Zb;    // bf16(h) after layers, Zb free by then

  // ---- convert weights to bf16 (per call; deterministic)
  auto cvt = [&](const float* in, unsigned short* o, size_t n) {
    cvt_f32_bf16_kernel<<<1024, 256, 0, stream>>>(in, o, n);
  };
  cvt(W1,  W1b,  (size_t)D_IN  * D_HID);
  cvt(W2,  W2b,  (size_t)D_HID * D_B);
  cvt(Wa,  Wab,  (size_t)NLAY * D_B * D_B);
  cvt(Wb,  Wbb,  (size_t)NLAY * D_B * D_B);
  cvt(Wd1, Wd1b, (size_t)D_B   * D_HID);
  cvt(Wd2, Wd2b, (size_t)D_HID * D_IN);

  const int gm = (N + BM - 1) / BM;
  dim3 gHid(gm, D_HID / BN);   // Nc = 512
  dim3 gB  (gm, D_B   / BN);   // Nc = 256
  dim3 gIn (gm, D_IN  / BN);   // Nc = 1024

  // ---- in_proj
  ln_in_kernel<<<N, 256, 0, stream>>>(x, lng, lnb, Xn);
  gemm_bf16_wmma<<<gHid, 256, 0, stream>>>(Xn, W1b, b1, nullptr, H1, N, D_IN, D_HID, 1);
  gemm_bf16_wmma<<<gB,   256, 0, stream>>>(H1, W2b, b2, h, nullptr, N, D_HID, D_B, 0);

  // ---- 3 GIN blocks
  for (int i = 0; i < NLAY; ++i) {
    (void)hipMemsetAsync(agg, 0, (size_t)N * D_B * 4, stream);
    scatter_add_kernel<<<(E + 3) / 4, 256, 0, stream>>>(src, dst, h, agg, E);
    add_cvt_bf16_kernel<<<2048, 256, 0, stream>>>(h, agg, Zb, (size_t)N * D_B);
    gemm_bf16_wmma<<<gB, 256, 0, stream>>>(Zb, Wab + (size_t)i * D_B * D_B,
                                           ba + i * D_B, nullptr, T1, N, D_B, D_B, 1);
    gemm_bf16_wmma<<<gB, 256, 0, stream>>>(T1, Wbb + (size_t)i * D_B * D_B,
                                           bb + i * D_B, Z2, nullptr, N, D_B, D_B, 0);
    ln_relu_residual_kernel<<<N, 256, 0, stream>>>(Z2, ng + i * D_B, nb + i * D_B, h);
  }

  // ---- delta_proj + output norm
  cvt_f32_bf16_kernel<<<2048, 256, 0, stream>>>(h, Hb, (size_t)N * D_B);
  gemm_bf16_wmma<<<gHid, 256, 0, stream>>>(Hb, Wd1b, bd1, nullptr, T2, N, D_B, D_HID, 1);
  gemm_bf16_wmma<<<gIn,  256, 0, stream>>>(T2, Wd2b, bd2, dlt, nullptr, N, D_HID, D_IN, 0);
  final_ln_kernel<<<N, 256, 0, stream>>>(x, dlt, outg, outb, out);
}
